// Pyg_SAGE_31104153158263
// MI455X (gfx1250) — compile-verified, matrix operand order checked by reading, not run on hardware
//
#include <hip/hip_runtime.h>
#include <hip/hip_bf16.h>
#include <math.h>

typedef __attribute__((ext_vector_type(16))) _Float16 v16h;
typedef __attribute__((ext_vector_type(8)))  _Float16 v8h;
typedef __attribute__((ext_vector_type(8)))  float    v8f;

#define N_NODES 50000
#define MP      50048          // rows padded to multiple of 128 (391 * 128)
#define EPS_NORM 1e-12f
#define EPS_BN   1e-5f

// ---------------------------------------------------------------------------
// Elementwise / graph kernels
// ---------------------------------------------------------------------------

__global__ void k_clip(const float* __restrict__ x, float* __restrict__ h, int n) {
    int i = blockIdx.x * blockDim.x + threadIdx.x;
    if (i < n) {
        float v = x[i];
        h[i] = fminf(0.4f, fmaxf(-0.4f, v));
    }
}

__global__ void k_deg(const int* __restrict__ dst, float* __restrict__ deg, int E) {
    int e = blockIdx.x * blockDim.x + threadIdx.x;
    if (e < E) atomicAdd(&deg[dst[e]], 1.0f);
}

// scatter: agg[dst] += h[src]   (one thread per (edge, feature))
__global__ void k_aggregate(const int* __restrict__ src, const int* __restrict__ dst,
                            const float* __restrict__ h, float* __restrict__ agg,
                            int E, int F) {
    long long i = (long long)blockIdx.x * blockDim.x + threadIdx.x;
    long long total = (long long)E * F;
    if (i < total) {
        int e = (int)(i / F);
        int f = (int)(i % F);
        atomicAdd(&agg[(size_t)dst[e] * F + f], h[(size_t)src[e] * F + f]);
    }
}

// ---------------------------------------------------------------------------
// Staging kernels (fp32 -> f16, padded layouts for WMMA)
// ---------------------------------------------------------------------------

// A = [ mean | self ]  in f16,  Mp x Kp rows,  Kp = 2*Kph, Kph = roundup32(Fin)
__global__ void k_stageA_conv(const float* __restrict__ agg, const float* __restrict__ deg,
                              const float* __restrict__ h, _Float16* __restrict__ A,
                              int Fin, int Kph) {
    int Kp = 2 * Kph;
    int i = blockIdx.x * blockDim.x + threadIdx.x;
    int total = MP * Kp;
    if (i >= total) return;
    int m = i / Kp, k = i % Kp;
    float v = 0.0f;
    if (m < N_NODES) {
        if (k < Fin)
            v = agg[(size_t)m * Fin + k] / fmaxf(deg[m], 1.0f);
        else if (k >= Kph && k < Kph + Fin)
            v = h[(size_t)m * Fin + (k - Kph)];
    }
    A[i] = (_Float16)v;
}

__global__ void k_stageA_single(const float* __restrict__ h, _Float16* __restrict__ A,
                                int Fin, int Kp) {
    int i = blockIdx.x * blockDim.x + threadIdx.x;
    int total = MP * Kp;
    if (i >= total) return;
    int m = i / Kp, k = i % Kp;
    float v = (m < N_NODES && k < Fin) ? h[(size_t)m * Fin + k] : 0.0f;
    A[i] = (_Float16)v;
}

// Bt[n][k] = concat(W_l, W_r)[k][n]  (column-major weights, f16, Np x Kp)
__global__ void k_stageB_conv(const float* __restrict__ Wl, const float* __restrict__ Wr,
                              _Float16* __restrict__ B, int Fin, int Fout, int Kph, int Np) {
    int Kp = 2 * Kph;
    int i = blockIdx.x * blockDim.x + threadIdx.x;
    int total = Np * Kp;
    if (i >= total) return;
    int n = i / Kp, k = i % Kp;
    float v = 0.0f;
    if (n < Fout) {
        if (k < Fin)
            v = Wl[(size_t)k * Fout + n];
        else if (k >= Kph && k < Kph + Fin)
            v = Wr[(size_t)(k - Kph) * Fout + n];
    }
    B[i] = (_Float16)v;
}

__global__ void k_stageB_single(const float* __restrict__ W, _Float16* __restrict__ B,
                                int Fin, int Fout, int Kp, int Np) {
    int i = blockIdx.x * blockDim.x + threadIdx.x;
    int total = Np * Kp;
    if (i >= total) return;
    int n = i / Kp, k = i % Kp;
    float v = (n < Fout && k < Fin) ? W[(size_t)k * Fout + n] : 0.0f;
    B[i] = (_Float16)v;
}

// ---------------------------------------------------------------------------
// WMMA GEMM: D[Mp x Np] = A[Mp x K] * Bt[Np x K]^T, f16 in / f32 out.
// 256-thread blocks = 8 waves; each wave computes a 16x64 output strip:
// 4 independent 16x16 accumulators share one A fragment per k-step
// (4x less A traffic, 4 independent back-to-back WMMAs keep the XDL pipe full).
// A-fragment per ISA 7.12.2 (16-bit A 16x32): lane m = lane&15,
// K base = (lane>=16)?8:0, halves {kb..kb+7} then {kb+16..kb+23}.
// B stored column-major (Bt row-major) so B-fragment load mirrors A.
// Requires K % 32 == 0 and Np % 64 == 0; grid = (Np/64, MP/128).
// ---------------------------------------------------------------------------
__global__ void k_wmma_gemm4(const _Float16* __restrict__ A, const _Float16* __restrict__ B,
                             float* __restrict__ D, int K, int Np) {
    int wave   = threadIdx.x >> 5;
    int lane   = threadIdx.x & 31;
    int tileM  = blockIdx.y * 8 + wave;
    int tileN0 = blockIdx.x * 4;
    int lm     = lane & 15;
    int kb     = (lane >= 16) ? 8 : 0;

    const _Float16* Ap  = A + (size_t)(tileM * 16 + lm) * K + kb;
    const _Float16* Bp0 = B + (size_t)(tileN0 * 16 + lm) * K + kb;

    v8f acc[4];
#pragma unroll
    for (int t = 0; t < 4; ++t)
#pragma unroll
        for (int i = 0; i < 8; ++i) acc[t][i] = 0.0f;

    for (int k0 = 0; k0 < K; k0 += 32) {
        v8h a0 = *(const v8h*)(Ap + k0);
        v8h a1 = *(const v8h*)(Ap + k0 + 16);
        v16h af;
#pragma unroll
        for (int i = 0; i < 8; ++i) { af[i] = a0[i]; af[i + 8] = a1[i]; }
#pragma unroll
        for (int t = 0; t < 4; ++t) {
            const _Float16* Bp = Bp0 + (size_t)t * 16 * K;
            v8h b0 = *(const v8h*)(Bp + k0);
            v8h b1 = *(const v8h*)(Bp + k0 + 16);
            v16h bf;
#pragma unroll
            for (int i = 0; i < 8; ++i) { bf[i] = b0[i]; bf[i + 8] = b1[i]; }
            acc[t] = __builtin_amdgcn_wmma_f32_16x16x32_f16(
                /*neg_a=*/false, af, /*neg_b=*/false, bf,
                /*c_mod=*/(short)0, acc[t], /*reuse_a=*/false, /*reuse_b=*/false);
        }
    }

    int rbase = tileM * 16 + ((lane >> 4) << 3);   // lanes 0-15: M=v, 16-31: M=v+8
#pragma unroll
    for (int t = 0; t < 4; ++t) {
        int col = (tileN0 + t) * 16 + lm;
#pragma unroll
        for (int v = 0; v < 8; ++v)
            D[(size_t)(rbase + v) * Np + col] = acc[t][v];
    }
}

// ---------------------------------------------------------------------------
// Post-ops
// ---------------------------------------------------------------------------

// conv epilogue: v = G + bias; row-L2 normalize; ReLU -> h (fp32, tight ld=Fout)
__global__ void k_post_conv(const float* __restrict__ G, const float* __restrict__ bias,
                            float* __restrict__ hout, int Fout, int Np) {
    __shared__ float red[256];
    int m = blockIdx.x;
    int f = threadIdx.x;                 // blockDim == 256
    float v = 0.0f;
    if (f < Fout) v = G[(size_t)m * Np + f] + bias[f];
    red[f] = v * v;
    __syncthreads();
    for (int s = 128; s > 0; s >>= 1) {
        if (f < s) red[f] += red[f + s];
        __syncthreads();
    }
    float scale = 1.0f / fmaxf(sqrtf(red[0]), EPS_NORM);
    if (f < Fout) hout[(size_t)m * Fout + f] = fmaxf(v * scale, 0.0f);
}

__global__ void k_bias_relu(const float* __restrict__ G, const float* __restrict__ b,
                            float* __restrict__ h, int F, int Np) {
    int i = blockIdx.x * blockDim.x + threadIdx.x;
    if (i >= N_NODES * F) return;
    int m = i / F, f = i % F;
    h[i] = fmaxf(G[(size_t)m * Np + f] + b[f], 0.0f);
}

__global__ void k_bias_bn_relu(const float* __restrict__ G, const float* __restrict__ b,
                               const float* __restrict__ gamma, const float* __restrict__ beta,
                               const float* __restrict__ mean,  const float* __restrict__ var,
                               float* __restrict__ h, int F, int Np) {
    int i = blockIdx.x * blockDim.x + threadIdx.x;
    if (i >= N_NODES * F) return;
    int m = i / F, f = i % F;
    float v = G[(size_t)m * Np + f] + b[f];
    v = (v - mean[f]) * rsqrtf(var[f] + EPS_BN) * gamma[f] + beta[f];
    h[i] = fmaxf(v, 0.0f);
}

// fc2b epilogue: logits = G + b, then row log-softmax over C=19 -> d_out
__global__ void k_logits_lsm(const float* __restrict__ G, const float* __restrict__ b,
                             float* __restrict__ out, int C, int Np) {
    __shared__ float sh[32];
    int m = blockIdx.x;
    int f = threadIdx.x;                 // blockDim == 32
    float v = (f < C) ? (G[(size_t)m * Np + f] + b[f]) : -3.0e38f;
    sh[f] = v;
    __syncthreads();
    for (int s = 16; s > 0; s >>= 1) { if (f < s) sh[f] = fmaxf(sh[f], sh[f + s]); __syncthreads(); }
    float mx = sh[0];
    __syncthreads();
    sh[f] = (f < C) ? __expf(v - mx) : 0.0f;
    __syncthreads();
    for (int s = 16; s > 0; s >>= 1) { if (f < s) sh[f] += sh[f + s]; __syncthreads(); }
    float lse = logf(sh[0]);
    if (f < C) out[(size_t)m * C + f] = v - mx - lse;
}

// ---------------------------------------------------------------------------
// Launcher
// ---------------------------------------------------------------------------
extern "C" void kernel_launch(void* const* d_in, const int* in_sizes, int n_in,
                              void* d_out, int out_size, void* d_ws, size_t ws_size,
                              hipStream_t stream) {
    (void)n_in; (void)out_size; (void)ws_size;

    const float* x      = (const float*)d_in[0];
    const int*   ei     = (const int*)d_in[1];
    const float* W1_l   = (const float*)d_in[2];
    const float* b1     = (const float*)d_in[3];
    const float* W1_r   = (const float*)d_in[4];
    const float* W2_l   = (const float*)d_in[5];
    const float* b2     = (const float*)d_in[6];
    const float* W2_r   = (const float*)d_in[7];
    const float* W3_l   = (const float*)d_in[8];
    const float* b3     = (const float*)d_in[9];
    const float* W3_r   = (const float*)d_in[10];
    const float* Wfc1   = (const float*)d_in[11];
    const float* bfc1   = (const float*)d_in[12];
    const float* Wfc2a  = (const float*)d_in[13];
    const float* bfc2a  = (const float*)d_in[14];
    const float* bng    = (const float*)d_in[15];
    const float* bnb    = (const float*)d_in[16];
    const float* bnm    = (const float*)d_in[17];
    const float* bnv    = (const float*)d_in[18];
    const float* Wfc2b  = (const float*)d_in[19];
    const float* bfc2b  = (const float*)d_in[20];
    float*       out    = (float*)d_out;

    const int E   = in_sizes[1] / 2;
    const int*  src = ei;
    const int*  dst = ei + E;

    // workspace carve-out (disjoint lifetimes: agg and G share OFF_AG slot)
    char* ws = (char*)d_ws;
    const size_t OFF_DEG = 0;                      // N_NODES f32
    const size_t OFF_H   = 0x40000;                // <= MP*256 f32  (feature buffer)
    const size_t OFF_AG  = OFF_H  + 0x3400000;     // agg (M*Fin f32) / GEMM out (MP*Np f32)
    const size_t OFF_AST = OFF_AG + 0x3400000;     // A stage f16, MP*448
    const size_t OFF_BST = OFF_AST + 0x2E00000;    // B stage f16, <= 256*448
    float*    deg  = (float*)(ws + OFF_DEG);
    float*    hbuf = (float*)(ws + OFF_H);
    float*    agg  = (float*)(ws + OFF_AG);
    float*    G    = (float*)(ws + OFF_AG);        // same slot, later lifetime
    _Float16* Ast  = (_Float16*)(ws + OFF_AST);
    _Float16* Bst  = (_Float16*)(ws + OFF_BST);

    const int TB = 256;
    auto blocks = [](long long n, int tb) { return (unsigned)((n + tb - 1) / tb); };
    const unsigned GEMM_GY = MP / 128;             // 391 (8 waves x 16 rows per block)

    // ---- degrees (per call; deterministic) ----
    hipMemsetAsync(deg, 0, (size_t)N_NODES * 4, stream);
    k_deg<<<blocks(E, TB), TB, 0, stream>>>(dst, deg, E);

    // ---- h0 = clip(x) ----
    k_clip<<<blocks((long long)N_NODES * 100, TB), TB, 0, stream>>>(x, hbuf, N_NODES * 100);

    // ---- SAGE conv layers (Np padded to multiple of 64 for 4-wide N blocking) ----
    struct ConvCfg { int Fin, Fout, Kph, Np; const float *Wl, *Wr, *b; };
    ConvCfg convs[3] = {
        {100, 200, 128, 256, W1_l, W1_r, b1},
        {200, 128, 224, 128, W2_l, W2_r, b2},
        {128, 128, 128, 128, W3_l, W3_r, b3},
    };
    for (int l = 0; l < 3; ++l) {
        const ConvCfg& c = convs[l];
        const int Kp = 2 * c.Kph;
        hipMemsetAsync(agg, 0, (size_t)N_NODES * c.Fin * 4, stream);
        k_aggregate<<<blocks((long long)E * c.Fin, TB), TB, 0, stream>>>(src, dst, hbuf, agg, E, c.Fin);
        k_stageA_conv<<<blocks((long long)MP * Kp, TB), TB, 0, stream>>>(agg, deg, hbuf, Ast, c.Fin, c.Kph);
        k_stageB_conv<<<blocks((long long)c.Np * Kp, TB), TB, 0, stream>>>(c.Wl, c.Wr, Bst, c.Fin, c.Fout, c.Kph, c.Np);
        k_wmma_gemm4<<<dim3(c.Np / 64, GEMM_GY), TB, 0, stream>>>(Ast, Bst, G, Kp, c.Np);
        k_post_conv<<<N_NODES, 256, 0, stream>>>(G, c.b, hbuf, c.Fout, c.Np);
    }

    // ---- fc1: relu(h @ Wfc1 + bfc1), 128 -> 128 ----
    {
        const int Fin = 128, Fout = 128, Kp = 128, Np = 128;
        k_stageA_single<<<blocks((long long)MP * Kp, TB), TB, 0, stream>>>(hbuf, Ast, Fin, Kp);
        k_stageB_single<<<blocks((long long)Np * Kp, TB), TB, 0, stream>>>(Wfc1, Bst, Fin, Fout, Kp, Np);
        k_wmma_gemm4<<<dim3(Np / 64, GEMM_GY), TB, 0, stream>>>(Ast, Bst, G, Kp, Np);
        k_bias_relu<<<blocks((long long)N_NODES * Fout, TB), TB, 0, stream>>>(G, bfc1, hbuf, Fout, Np);
    }

    // ---- fc2a + BN + relu: 128 -> 256 ----
    {
        const int Fin = 128, Fout = 256, Kp = 128, Np = 256;
        k_stageA_single<<<blocks((long long)MP * Kp, TB), TB, 0, stream>>>(hbuf, Ast, Fin, Kp);
        k_stageB_single<<<blocks((long long)Np * Kp, TB), TB, 0, stream>>>(Wfc2a, Bst, Fin, Fout, Kp, Np);
        k_wmma_gemm4<<<dim3(Np / 64, GEMM_GY), TB, 0, stream>>>(Ast, Bst, G, Kp, Np);
        k_bias_bn_relu<<<blocks((long long)N_NODES * Fout, TB), TB, 0, stream>>>(
            G, bfc2a, bng, bnb, bnm, bnv, hbuf, Fout, Np);
    }

    // ---- fc2b + log_softmax: 256 -> 19 (Np padded to 64) ----
    {
        const int Fin = 256, Fout = 19, Kp = 256, Np = 64;
        k_stageA_single<<<blocks((long long)MP * Kp, TB), TB, 0, stream>>>(hbuf, Ast, Fin, Kp);
        k_stageB_single<<<blocks((long long)Np * Kp, TB), TB, 0, stream>>>(Wfc2b, Bst, Fin, Fout, Kp, Np);
        k_wmma_gemm4<<<dim3(Np / 64, GEMM_GY), TB, 0, stream>>>(Ast, Bst, G, Kp, Np);
        k_logits_lsm<<<N_NODES, 32, 0, stream>>>(G, bfc2b, out, Fout, Np);
    }
}
